// VSSBlock_34059090657705
// MI455X (gfx1250) — matrix operand order repaired and checked by reading, not compiled, not run on hardware
//
#include <hip/hip_runtime.h>
#include <hip/hip_bf16.h>

// ---------------- problem constants ----------------
#define BATCH 4
#define HH    64
#define WW    64
#define CC    192
#define DI    384          // 2*C
#define L2    1024         // (H/2)*(W/2)
#define K4    4
#define NS    16
#define RR    24           // ceil(DI/16)
#define CDBL  56           // R + 2N
#define MTOT  (BATCH*HH*WW) // 16384 positions

typedef __attribute__((ext_vector_type(16))) __bf16 v16bf;
typedef __attribute__((ext_vector_type(8)))  float  v8f;

__device__ __forceinline__ unsigned short f2bf(float x) {
  unsigned int u = __float_as_uint(x);
  u += 0x7FFFu + ((u >> 16) & 1u);       // round-to-nearest-even
  return (unsigned short)(u >> 16);
}
__device__ __forceinline__ float wave_sum(float v) {
  #pragma unroll
  for (int o = 16; o > 0; o >>= 1) v += __shfl_xor(v, o, 32);
  return v;
}
__device__ __forceinline__ float silu_f(float v) { return v / (1.f + __expf(-v)); }

// ============ WMMA 2x2 macro-tile: 32x32 output per wave, each fragment feeds 2 WMMAs ======
// A lane layout:  lane l holds row m=l&15; e<8 -> k=kk+8*(l>>4)+e ; e>=8 -> k=kk+16+8*(l>>4)+(e-8)
// B lane layout:  lane l holds column n=l&15; element e -> row k, same k order as A
struct Acc2x2 { v8f m0n0, m0n1, m1n0, m1n1; };

__device__ __forceinline__ Acc2x2 wmma_2x2(const unsigned short* __restrict__ A, int lda,
                                           const unsigned short* __restrict__ Wt, int ldn,
                                           int rowBase, int colBase, int Ktot) {
  const int lane = threadIdx.x & 31;
  const int half = lane >> 4;
  const int mr   = lane & 15;
  v8f zz = {0.f,0.f,0.f,0.f,0.f,0.f,0.f,0.f};
  Acc2x2 acc = {zz, zz, zz, zz};
  const unsigned short* arow0 = A + (size_t)(rowBase + mr) * lda;
  const unsigned short* arow1 = arow0 + (size_t)16 * lda;
  for (int kk = 0; kk < Ktot; kk += 32) {
    union { v16bf v; uint4 q[2]; } a0, a1, b0, b1;
    const uint4* pa0 = (const uint4*)(arow0 + kk + 8*half);
    const uint4* pa1 = (const uint4*)(arow1 + kk + 8*half);
    a0.q[0] = pa0[0]; a0.q[1] = pa0[2];          // k = kk+8h..+7 , kk+16+8h..+7
    a1.q[0] = pa1[0]; a1.q[1] = pa1[2];
    const unsigned short* pb = Wt + colBase + mr;
    unsigned short* b0s = (unsigned short*)&b0;
    unsigned short* b1s = (unsigned short*)&b1;
    #pragma unroll
    for (int s = 0; s < 16; s++) {
      const int krow = kk + ((s < 8) ? (8*half + s) : (16 + 8*half + (s - 8)));
      b0s[s] = pb[(size_t)krow * ldn];
      b1s[s] = pb[(size_t)krow * ldn + 16];
    }
    acc.m0n0 = __builtin_amdgcn_wmma_f32_16x16x32_bf16(false, a0.v, false, b0.v, (short)0, acc.m0n0, false, false);
    acc.m0n1 = __builtin_amdgcn_wmma_f32_16x16x32_bf16(false, a0.v, false, b1.v, (short)0, acc.m0n1, false, false);
    acc.m1n0 = __builtin_amdgcn_wmma_f32_16x16x32_bf16(false, a1.v, false, b0.v, (short)0, acc.m1n0, false, false);
    acc.m1n1 = __builtin_amdgcn_wmma_f32_16x16x32_bf16(false, a1.v, false, b1.v, (short)0, acc.m1n1, false, false);
  }
  return acc;
}

__device__ __forceinline__ void store_tile_f32(float* __restrict__ D, int ldd,
                                               int rowBase, int colBase, v8f a) {
  int lane = threadIdx.x & 31, half = lane >> 4, mr = lane & 15;
  #pragma unroll
  for (int r = 0; r < 8; r++)
    D[(size_t)(rowBase + r + 8*half) * ldd + colBase + mr] = a[r];
}

// ---------------- weight prep: W[Nrows][Kcols] f32 -> Wt[Kpad][Npad] bf16 (zero pad) --------
__global__ void k_transpose_w(const float* __restrict__ W, unsigned short* __restrict__ Wt,
                              int Nrows, int Kcols, int Npad, int Kpad) {
  int i = blockIdx.x * blockDim.x + threadIdx.x;
  if (i >= Kpad * Npad) return;
  int kr = i / Npad, n = i % Npad;
  float v = (kr < Kcols && n < Nrows) ? W[(size_t)n * Kcols + kr] : 0.f;
  Wt[i] = f2bf(v);
}

// ---------------- LN over C=192, wave per row, bf16 out -------------------------------------
__global__ void k_ln1(const float* __restrict__ x, const float* __restrict__ w,
                      const float* __restrict__ b, unsigned short* __restrict__ xln) {
  int row  = blockIdx.x * blockDim.y + threadIdx.y;
  int lane = threadIdx.x;
  const float* xr = x + (size_t)row * CC;
  float v[6], s = 0.f;
  #pragma unroll
  for (int i = 0; i < 6; i++) { v[i] = xr[lane + 32*i]; s += v[i]; }
  s = wave_sum(s);
  float mu = s * (1.f/CC), q = 0.f;
  #pragma unroll
  for (int i = 0; i < 6; i++) { float d = v[i]-mu; q += d*d; }
  q = wave_sum(q);
  float rs = rsqrtf(q * (1.f/CC) + 1e-6f);
  unsigned short* o = xln + (size_t)row * CC;
  #pragma unroll
  for (int i = 0; i < 6; i++) {
    int c = lane + 32*i;
    o[c] = f2bf((v[i]-mu)*rs*w[c] + b[c]);
  }
}

// ---------------- in_proj: xln(16384x192) @ Wt(192x768) -> x(f32) | z(f32), uniform split ---
__global__ void k_gemm_inproj(const unsigned short* __restrict__ xln,
                              const unsigned short* __restrict__ Wt,
                              float* __restrict__ xconv, float* __restrict__ z) {
  int wave = threadIdx.x >> 5;
  int tileM2 = blockIdx.y*4 + wave;        // 32-row macro tile
  int tileN2 = blockIdx.x;                 // 32-col macro tile (0..23)
  int rowBase = tileM2 * 32, colBase = tileN2 * 32;
  Acc2x2 acc = wmma_2x2(xln, CC, Wt, 2*DI, rowBase, colBase, CC);
  // wave-uniform destination select: cols [0,384) -> xconv, [384,768) -> z
  float* dst = (colBase < DI) ? (xconv + colBase) : (z + (colBase - DI));
  store_tile_f32(dst, DI, rowBase,      0, acc.m0n0);
  store_tile_f32(dst, DI, rowBase,     16, acc.m0n1);
  store_tile_f32(dst, DI, rowBase + 16, 0, acc.m1n0);
  store_tile_f32(dst, DI, rowBase + 16,16, acc.m1n1);
}

// ---------------- depthwise 3x3 conv + bias + SiLU, scatter to 4-direction layout -----------
__global__ void k_conv(const float* __restrict__ xc, const float* __restrict__ cw,
                       const float* __restrict__ cb,
                       unsigned short* __restrict__ xs_bf, float* __restrict__ u) {
  size_t idx = (size_t)blockIdx.x * blockDim.x + threadIdx.x;
  if (idx >= (size_t)MTOT * DI) return;
  int d  = idx % DI;
  size_t p = idx / DI;
  int wp = p % WW, hp = (p / WW) % HH, b = p / (WW*HH);
  float s = 0.f;
  #pragma unroll
  for (int r = 0; r < 3; r++) {
    int hh = hp + r - 1;
    if (hh < 0 || hh >= HH) continue;
    #pragma unroll
    for (int c = 0; c < 3; c++) {
      int wc = wp + c - 1;
      if (wc < 0 || wc >= WW) continue;
      s += xc[(((size_t)b*HH + hh)*WW + wc)*DI + d] * cw[d*9 + r*3 + c];
    }
  }
  s = silu_f(s + cb[d]);
  int kr = hp & 1, kc = wp & 1;
  int k  = kr ? (kc ? 1 : 3) : (kc ? 2 : 0);  // (0,0)->0 (1,1)->1 (0,1)->2 (1,0)->3
  int l  = (hp>>1)*32 + (wp>>1);
  int bk = b*4 + k;
  xs_bf[((size_t)bk*L2 + l)*DI + d] = f2bf(s);   // A for x_proj GEMM  [pos][d]
  u    [((size_t)bk*DI + d)*L2 + l] = s;         // scan u             [bk][d][l]
}

// ---------------- x_proj: xs(1024x384) @ Wt_k(384x64) -> x_dbl f32 + dt-input bf16 ----------
__global__ void k_gemm_xproj(const unsigned short* __restrict__ xs_bf,
                             const unsigned short* __restrict__ WtXp,
                             float* __restrict__ xdbl, unsigned short* __restrict__ dtsA) {
  int wave = threadIdx.x >> 5;
  int tileM2 = blockIdx.y*4 + wave;        // 0..31
  int tileN2 = blockIdx.x;                 // 0..1
  int bk = blockIdx.z;
  const unsigned short* A  = xs_bf + (size_t)bk*L2*DI;
  const unsigned short* Wt = WtXp  + (size_t)(bk&3)*DI*64;
  int rowBase = tileM2*32, colBase = tileN2*32;
  Acc2x2 acc = wmma_2x2(A, DI, Wt, 64, rowBase, colBase, DI);
  int lane = threadIdx.x & 31, half = lane>>4, mr = lane&15;
  auto epi = [&](v8f a, int rb, int cb) {
    #pragma unroll
    for (int r = 0; r < 8; r++) {
      int l = rb + r + 8*half;
      int c = cb + mr;
      size_t row = (size_t)bk*L2 + l;
      float vv = a[r];
      if (c < CDBL) xdbl[row*CDBL + c] = vv;
      if (c < 32)   dtsA[row*32 + c] = (c < RR) ? f2bf(vv) : (unsigned short)0; // K-pad zeros
    }
  };
  epi(acc.m0n0, rowBase,      colBase);
  epi(acc.m0n1, rowBase,      colBase + 16);
  epi(acc.m1n0, rowBase + 16, colBase);
  epi(acc.m1n1, rowBase + 16, colBase + 16);
}

// ---------------- dt: dtsA(1024x32) @ Wt_k(32x384) -> softplus(.+dt_b) in [bk][d][l] --------
__global__ void k_gemm_dt(const unsigned short* __restrict__ dtsA,
                          const unsigned short* __restrict__ WtDt,
                          const float* __restrict__ dt_b, float* __restrict__ dt) {
  int wave = threadIdx.x >> 5;
  int tileM2 = blockIdx.y*4 + wave;        // 0..31
  int tileN2 = blockIdx.x;                 // 0..11
  int bk = blockIdx.z;
  int k = bk & 3;
  const unsigned short* A  = dtsA + (size_t)bk*L2*32;
  const unsigned short* Wt = WtDt + (size_t)k*32*DI;
  int rowBase = tileM2*32, colBase = tileN2*32;
  Acc2x2 acc = wmma_2x2(A, 32, Wt, DI, rowBase, colBase, 32);
  int lane = threadIdx.x & 31, half = lane>>4, mr = lane&15;
  auto epi = [&](v8f a, int rb, int cb) {
    int d = cb + mr;
    float bias = dt_b[k*DI + d];
    #pragma unroll
    for (int r = 0; r < 8; r++) {
      int l = rb + r + 8*half;
      float vv = a[r] + bias;
      float sp = (vv > 20.f) ? vv : log1pf(__expf(vv));
      dt[((size_t)bk*DI + d)*L2 + l] = sp;
    }
  };
  epi(acc.m0n0, rowBase,      colBase);
  epi(acc.m0n1, rowBase,      colBase + 16);
  epi(acc.m1n0, rowBase + 16, colBase);
  epi(acc.m1n1, rowBase + 16, colBase + 16);
}

// ---------------- selective scan: 16 blocks x 384 threads (one (b,k,d) channel each) --------
__global__ void __launch_bounds__(384)
k_scan(const float* __restrict__ dt, const float* __restrict__ u,
       const float* __restrict__ xdbl, const float* __restrict__ A_logs,
       const float* __restrict__ Ds, float* __restrict__ ysp) {
  __shared__ float bc[128*32];           // 128 steps x (B[16] | C[16])
  int bk = blockIdx.x;
  int b = bk >> 2, k = bk & 3;
  int d = threadIdx.x;
  float Av[NS];
  #pragma unroll
  for (int n = 0; n < NS; n++) Av[n] = -__expf(A_logs[(size_t)(k*DI + d)*NS + n]);
  float Dv = Ds[k*DI + d];
  const float* dtp = dt + ((size_t)bk*DI + d)*L2;
  const float* up  = u  + ((size_t)bk*DI + d)*L2;
  float h[NS];
  #pragma unroll
  for (int n = 0; n < NS; n++) h[n] = 0.f;
  int kr = (k==1 || k==3), kc = (k==1 || k==2);
  for (int l0 = 0; l0 < L2; l0 += 128) {
    // prefetch next chunk of this channel's dt/u streams (global_prefetch_b8)
    if (l0 + 128 < L2) {
      __builtin_prefetch(dtp + l0 + 128, 0, 1);
      __builtin_prefetch(up  + l0 + 128, 0, 1);
    }
    __syncthreads();
    for (int idx = threadIdx.x; idx < 128*32; idx += 384) {
      int st = idx >> 5, j = idx & 31;
      bc[idx] = xdbl[((size_t)bk*L2 + l0 + st)*CDBL + RR + j];
    }
    __syncthreads();
    for (int t = 0; t < 128; t++) {
      int l = l0 + t;
      float dtv = dtp[l], uv = up[l];
      float dtu = dtv * uv;
      const float* Bv = &bc[t*32];
      float y = 0.f;
      #pragma unroll
      for (int n = 0; n < NS; n++) {
        h[n] = __expf(dtv * Av[n]) * h[n] + dtu * Bv[n];
        y += h[n] * Bv[16 + n];
      }
      y += Dv * uv;
      int hp = 2*(l>>5) + kr, wp = 2*(l&31) + kc;
      ysp[(((size_t)b*HH + hp)*WW + wp)*DI + d] = y;     // [b][h][w][d] for out_norm
    }
  }
}

// ---------------- residual proj: xln(16384x192) @ Wt(192x384) -> r f32 ----------------------
__global__ void k_gemm_res(const unsigned short* __restrict__ xln,
                           const unsigned short* __restrict__ Wt,
                           float* __restrict__ rbuf) {
  int wave = threadIdx.x >> 5;
  int tileM2 = blockIdx.y*4 + wave;
  int tileN2 = blockIdx.x;                 // 0..11
  int rowBase = tileM2*32, colBase = tileN2*32;
  Acc2x2 acc = wmma_2x2(xln, CC, Wt, DI, rowBase, colBase, CC);
  store_tile_f32(rbuf, DI, rowBase,      colBase,      acc.m0n0);
  store_tile_f32(rbuf, DI, rowBase,      colBase + 16, acc.m0n1);
  store_tile_f32(rbuf, DI, rowBase + 16, colBase,      acc.m1n0);
  store_tile_f32(rbuf, DI, rowBase + 16, colBase + 16, acc.m1n1);
}

// ---------------- out_norm LN + residual add + SiLU gate -> bf16 A for out_proj -------------
__global__ void k_fuse(const float* __restrict__ ysp, const float* __restrict__ onw,
                       const float* __restrict__ onb, const float* __restrict__ rbuf,
                       const float* __restrict__ z, unsigned short* __restrict__ yf) {
  int row  = blockIdx.x * blockDim.y + threadIdx.y;
  int lane = threadIdx.x;
  const float* yr = ysp + (size_t)row * DI;
  float v[12], s = 0.f;
  #pragma unroll
  for (int i = 0; i < 12; i++) { v[i] = yr[lane + 32*i]; s += v[i]; }
  s = wave_sum(s);
  float mu = s * (1.f/DI), q = 0.f;
  #pragma unroll
  for (int i = 0; i < 12; i++) { float dd = v[i]-mu; q += dd*dd; }
  q = wave_sum(q);
  float rs = rsqrtf(q * (1.f/DI) + 1e-5f);
  #pragma unroll
  for (int i = 0; i < 12; i++) {
    int c = lane + 32*i;
    float val = (v[i]-mu)*rs*onw[c] + onb[c] + rbuf[(size_t)row*DI + c];
    float zv  = z[(size_t)row*DI + c];
    yf[(size_t)row*DI + c] = f2bf(val * silu_f(zv));
  }
}

// ---------------- out_proj: yf(16384x384) @ Wt(384x192) + inputs -> d_out -------------------
__global__ void k_gemm_out(const unsigned short* __restrict__ yf,
                           const unsigned short* __restrict__ Wt,
                           const float* __restrict__ inp, float* __restrict__ out) {
  int wave = threadIdx.x >> 5;
  int tileM2 = blockIdx.y*4 + wave;
  int tileN2 = blockIdx.x;                 // 0..5
  int rowBase = tileM2*32, colBase = tileN2*32;
  Acc2x2 acc = wmma_2x2(yf, DI, Wt, CC, rowBase, colBase, DI);
  int lane = threadIdx.x & 31, half = lane>>4, mr = lane&15;
  auto epi = [&](v8f a, int rb, int cb) {
    #pragma unroll
    for (int r = 0; r < 8; r++) {
      size_t o = (size_t)(rb + r + 8*half) * CC + cb + mr;
      out[o] = inp[o] + a[r];
    }
  };
  epi(acc.m0n0, rowBase,      colBase);
  epi(acc.m0n1, rowBase,      colBase + 16);
  epi(acc.m1n0, rowBase + 16, colBase);
  epi(acc.m1n1, rowBase + 16, colBase + 16);
}

// =====================================================================================
extern "C" void kernel_launch(void* const* d_in, const int* in_sizes, int n_in,
                              void* d_out, int out_size, void* d_ws, size_t ws_size,
                              hipStream_t stream) {
  const float* inputs     = (const float*)d_in[0];
  const float* ln1_w      = (const float*)d_in[1];
  const float* ln1_b      = (const float*)d_in[2];
  const float* in_proj_w  = (const float*)d_in[3];
  const float* conv_w     = (const float*)d_in[4];
  const float* conv_b     = (const float*)d_in[5];
  const float* x_proj_w   = (const float*)d_in[6];
  const float* dt_w       = (const float*)d_in[7];
  const float* dt_b       = (const float*)d_in[8];
  const float* A_logs     = (const float*)d_in[9];
  const float* Ds         = (const float*)d_in[10];
  const float* out_norm_w = (const float*)d_in[11];
  const float* out_norm_b = (const float*)d_in[12];
  const float* res_proj_w = (const float*)d_in[13];
  const float* out_proj_w = (const float*)d_in[14];
  float* out = (float*)d_out;

  // ---- workspace carve-up (256B aligned) ----
  char* ws = (char*)d_ws;
  size_t off = 0;
  auto take = [&](size_t bytes) { char* p = ws + off; off += (bytes + 255) & ~(size_t)255; return p; };
  unsigned short* xln   = (unsigned short*)take((size_t)MTOT*CC*2);
  unsigned short* WtIn  = (unsigned short*)take((size_t)CC*2*DI*2);        // 192x768
  unsigned short* WtXp  = (unsigned short*)take((size_t)K4*DI*64*2);       // 4x(384x64)
  unsigned short* WtDt  = (unsigned short*)take((size_t)K4*32*DI*2);       // 4x(32x384)
  unsigned short* WtRes = (unsigned short*)take((size_t)CC*DI*2);          // 192x384
  unsigned short* WtOut = (unsigned short*)take((size_t)DI*CC*2);          // 384x192
  float*          xconv = (float*)take((size_t)MTOT*DI*4);                 // reused as rbuf
  float*          zbuf  = (float*)take((size_t)MTOT*DI*4);
  unsigned short* xs_bf = (unsigned short*)take((size_t)MTOT*DI*2);        // reused as yf
  float*          ubuf  = (float*)take((size_t)MTOT*DI*4);
  float*          xdbl  = (float*)take((size_t)MTOT*CDBL*4);
  unsigned short* dtsA  = (unsigned short*)take((size_t)MTOT*32*2);
  float*          dtbuf = (float*)take((size_t)MTOT*DI*4);
  float*          ysp   = (float*)take((size_t)MTOT*DI*4);
  float*          rbuf  = xconv;          // dead after conv
  unsigned short* yf    = xs_bf;          // dead after x_proj GEMM

  auto blks = [](int total) { return dim3((total + 255) / 256); };

  // ---- weight prep (bf16, transposed to [K][Npad]) ----
  k_transpose_w<<<blks(CC*2*DI),   256, 0, stream>>>(in_proj_w,  WtIn,  2*DI, CC, 2*DI, CC);
  for (int k = 0; k < K4; k++)
    k_transpose_w<<<blks(DI*64),   256, 0, stream>>>(x_proj_w + (size_t)k*CDBL*DI,
                                                     WtXp + (size_t)k*DI*64, CDBL, DI, 64, DI);
  for (int k = 0; k < K4; k++)
    k_transpose_w<<<blks(32*DI),   256, 0, stream>>>(dt_w + (size_t)k*DI*RR,
                                                     WtDt + (size_t)k*32*DI, DI, RR, DI, 32);
  k_transpose_w<<<blks(CC*DI),     256, 0, stream>>>(res_proj_w, WtRes, DI, CC, DI, CC);
  k_transpose_w<<<blks(DI*CC),     256, 0, stream>>>(out_proj_w, WtOut, CC, DI, CC, DI);

  // ---- pipeline (2x2 macro-tiles: grid.x = Ntiles/2, grid.y = Mtiles/2/4waves) ----
  k_ln1        <<<dim3(MTOT/8),  dim3(32,8), 0, stream>>>(inputs, ln1_w, ln1_b, xln);
  k_gemm_inproj<<<dim3(24, 128), dim3(128),  0, stream>>>(xln, WtIn, xconv, zbuf);
  k_conv       <<<blks(MTOT*DI), 256,        0, stream>>>(xconv, conv_w, conv_b, xs_bf, ubuf);
  k_gemm_xproj <<<dim3(2, 8, 16),  dim3(128), 0, stream>>>(xs_bf, WtXp, xdbl, dtsA);
  k_gemm_dt    <<<dim3(12, 8, 16), dim3(128), 0, stream>>>(dtsA, WtDt, dt_b, dtbuf);
  k_scan       <<<dim3(16),      dim3(384),  0, stream>>>(dtbuf, ubuf, xdbl, A_logs, Ds, ysp);
  k_gemm_res   <<<dim3(12, 128), dim3(128),  0, stream>>>(xln, WtRes, rbuf);
  k_fuse       <<<dim3(MTOT/8),  dim3(32,8), 0, stream>>>(ysp, out_norm_w, out_norm_b, rbuf, zbuf, yf);
  k_gemm_out   <<<dim3(6, 128),  dim3(128),  0, stream>>>(yf, WtOut, inputs, out);
}
